// PAMCmplx_Module_30812095382084
// MI455X (gfx1250) — matrix operand order repaired and checked by reading, not compiled, not run on hardware
//
#include <hip/hip_runtime.h>
#include <hip/hip_bf16.h>

// ---------------------------------------------------------------------------
// PAMCmplx fused complex attention for gfx1250 (MI455X), wave32 + WMMA bf16.
//   B=2, CG=256 complex chans, L=4096, DQK=8
// Pipeline:
//   K0: transpose+convert x -> xbT[comp][b][l][c] bf16 (LDS tile transpose,
//       async global->LDS when available)
//   K0b: wv -> bf16
//   K1: Q/K complex projections, packed bf16 [qr|qi], [kr|-ki], [ki|kr]
//   K2: V complex projection via v_wmma_f32_16x16x32_bf16 (RR/II/RI/IR),
//       all operands contiguous bf16 loads
//   K3: fused two-pass softmax attention; key rows staged in LDS (async),
//       P tiles staged in LDS, P x V^T via WMMA, out = gamma*acc + x
// ---------------------------------------------------------------------------

typedef __bf16 v16bf __attribute__((ext_vector_type(16)));
typedef __bf16 v8bf  __attribute__((ext_vector_type(8)));
typedef float  v8f   __attribute__((ext_vector_type(8)));
typedef int    v4i_  __attribute__((ext_vector_type(4)));

#define B_   2
#define CG_  256
#define L_   4096
#define D_   8
#define CTOT 512

#define WMMA_BF16(A, Bm, Cm) \
  __builtin_amdgcn_wmma_f32_16x16x32_bf16(false, (A), false, (Bm), (short)0, (Cm), false, false)

#if defined(__gfx1250__) && \
    __has_builtin(__builtin_amdgcn_global_load_async_to_lds_b128) && \
    __has_builtin(__builtin_amdgcn_s_wait_asynccnt)
#define USE_ASYNC_LDS 1
#endif

#ifdef USE_ASYNC_LDS
#define GLOBAL_AS __attribute__((address_space(1)))
#define LDS_AS    __attribute__((address_space(3)))
// builtin signature (from toolchain diagnostic): (int4 AS1*, int4 AS3*, Ii, Ii)
#define ASYNC_CP16(gsrc, ldst) \
  __builtin_amdgcn_global_load_async_to_lds_b128( \
      (GLOBAL_AS v4i_*)(void*)(size_t)(const void*)(gsrc), \
      (LDS_AS v4i_*)(void*)(ldst), 0, 0)
#endif

static __device__ __forceinline__ v8f v8f_zero() {
  v8f z;
#pragma unroll
  for (int i = 0; i < 8; ++i) z[i] = 0.0f;
  return z;
}

static __device__ __forceinline__ v16bf bf_cat(v8bf lo, v8bf hi) {
  v16bf r;
#pragma unroll
  for (int i = 0; i < 8; ++i) { r[i] = lo[i]; r[i + 8] = hi[i]; }
  return r;
}

// ---------------------------------------------------------------------------
// K0: x (fp32, [comp][b][c][l]) -> xbT (bf16, [comp][b][l][c]) via LDS tile.
// ---------------------------------------------------------------------------
__global__ __launch_bounds__(256)
void xpose_kernel(const float* __restrict__ x, __bf16* __restrict__ xbT) {
  __shared__ __align__(16) float tile[32][36];   // 36: keep 16B align + de-conflict

  const int t   = threadIdx.x;
  const int blk = blockIdx.x;            // (comp*B+b) * 8 * 128
  const int cb  = blk >> 10;             // 0..3
  const int r   = blk & 1023;
  const int c0  = (r >> 7) * 32;
  const int l0  = (r & 127) * 32;
  const float* src = x + ((size_t)cb * CG_) * L_;

#ifdef USE_ASYNC_LDS
  {
    const int row = t >> 3;              // channel row 0..31
    const int seg = (t & 7) * 4;         // 4 floats = 16B per thread
    ASYNC_CP16(src + (size_t)(c0 + row) * L_ + l0 + seg, &tile[row][seg]);
  }
  __builtin_amdgcn_s_wait_asynccnt(0);
#else
#pragma unroll
  for (int it = 0; it < 4; ++it) {
    const int row = (t >> 5) + it * 8;
    const int col = t & 31;
    tile[row][col] = src[(size_t)(c0 + row) * L_ + l0 + col];
  }
#endif
  __syncthreads();

#pragma unroll
  for (int it = 0; it < 4; ++it) {
    const int lr = (t >> 5) + it * 8;    // l row in tile
    const int cc = t & 31;               // channel column
    xbT[((size_t)cb * L_ + l0 + lr) * CG_ + c0 + cc] = (__bf16)tile[cc][lr];
  }
}

// ---------------------------------------------------------------------------
// K0b: wv fp32 -> bf16 (131072 elements)
// ---------------------------------------------------------------------------
__global__ __launch_bounds__(256)
void wconv_kernel(const float* __restrict__ wv, __bf16* __restrict__ wvb) {
  const int i = blockIdx.x * 256 + threadIdx.x;
  wvb[i] = (__bf16)wv[i];
}

// ---------------------------------------------------------------------------
// K1: Q/K complex 1x1 conv + packing (reads bf16 xbT sequentially).
//   qa[b][l][0..7]=qr,[8..15]=qi ; ker=[kr|-ki] ; kei=[ki|kr]
// ---------------------------------------------------------------------------
__global__ __launch_bounds__(256)
void qk_pack_kernel(const __bf16* __restrict__ xbT,
                    const float* __restrict__ wq,
                    const float* __restrict__ wk,
                    __bf16* __restrict__ qa,
                    __bf16* __restrict__ ker,
                    __bf16* __restrict__ kei) {
  const int idx = blockIdx.x * blockDim.x + threadIdx.x;  // b*8*L + d*L + l
  const int b = idx >> 15;
  const int r = idx & 32767;
  const int d = r >> 12;
  const int l = r & 4095;

  const __bf16* xrp = xbT + ((size_t)(0 * B_ + b) * L_ + l) * CG_;
  const __bf16* xip = xbT + ((size_t)(1 * B_ + b) * L_ + l) * CG_;
  const float* wqr = wq + d * CG_;
  const float* wqi = wq + D_ * CG_ + d * CG_;
  const float* wkr = wk + d * CG_;
  const float* wki = wk + D_ * CG_ + d * CG_;

  float qr = 0.f, qi = 0.f, kr = 0.f, ki = 0.f;
  for (int c = 0; c < CG_; ++c) {
    const float xrv = (float)xrp[c];
    const float xiv = (float)xip[c];
    const float aq = wqr[c], bq = wqi[c];
    qr += aq * xrv - bq * xiv;
    qi += aq * xiv + bq * xrv;
    const float ak = wkr[c], bk = wki[c];
    kr += ak * xrv - bk * xiv;
    ki += ak * xiv + bk * xrv;
  }
  const size_t base = ((size_t)(b * L_ + l)) << 4;
  qa [base + d]     = (__bf16)qr;
  qa [base + 8 + d] = (__bf16)qi;
  ker[base + d]     = (__bf16)kr;
  ker[base + 8 + d] = (__bf16)(-ki);
  kei[base + d]     = (__bf16)ki;
  kei[base + 8 + d] = (__bf16)kr;
}

// ---------------------------------------------------------------------------
// K2: complex V GEMM, one wave per 16x16 tile; all-bf16 contiguous loads.
// ---------------------------------------------------------------------------
__global__ __launch_bounds__(32)
void vgemm_kernel(const __bf16* __restrict__ xbT,
                  const __bf16* __restrict__ wvb,
                  __bf16* __restrict__ vbuf) {
  const int lane = threadIdx.x & 31;
  const int half = lane >> 4;
  const int n    = lane & 15;

  const int tile = blockIdx.x;         // B * 16 * 256
  const int b  = tile >> 12;
  const int ct = (tile >> 8) & 15;
  const int lt = tile & 255;
  const int c0 = ct * 16;
  const int l0 = lt * 16;

  const __bf16* xrRow = xbT + ((size_t)(0 * B_ + b) * L_ + l0 + n) * CG_;
  const __bf16* xiRow = xbT + ((size_t)(1 * B_ + b) * L_ + l0 + n) * CG_;
  const __bf16* wRow  = wvb + (size_t)(c0 + n) * CG_;

  v8f RR = v8f_zero(), II = v8f_zero(), RI = v8f_zero(), IR = v8f_zero();

  for (int kk = 0; kk < CG_; kk += 32) {
    // A tiles: W row = c0+n, K = kk + half*16 + e (contiguous)
    const __bf16* wr = wRow + kk + half * 16;
    const __bf16* wi = wr + (size_t)CG_ * CG_;
    const v16bf a_r = bf_cat(*(const v8bf*)wr, *(const v8bf*)(wr + 8));
    const v16bf a_i = bf_cat(*(const v8bf*)wi, *(const v8bf*)(wi + 8));
    // B tiles: col = l0+n, K = kk + half*16 + e (contiguous in xbT)
    const __bf16* xr = xrRow + kk + half * 16;
    const __bf16* xi = xiRow + kk + half * 16;
    const v16bf b_r = bf_cat(*(const v8bf*)xr, *(const v8bf*)(xr + 8));
    const v16bf b_i = bf_cat(*(const v8bf*)xi, *(const v8bf*)(xi + 8));

    RR = WMMA_BF16(a_r, b_r, RR);
    II = WMMA_BF16(a_i, b_i, II);
    RI = WMMA_BF16(a_r, b_i, RI);
    IR = WMMA_BF16(a_i, b_r, IR);
  }

#pragma unroll
  for (int i = 0; i < 8; ++i) {
    const int c = c0 + i + half * 8;
    const int l = l0 + n;
    vbuf[((size_t)b * CTOT + c) * L_ + l]       = (__bf16)(RR[i] - II[i]);
    vbuf[((size_t)b * CTOT + 256 + c) * L_ + l] = (__bf16)(RI[i] + IR[i]);
  }
}

// ---------------------------------------------------------------------------
// K3: fused attention. 1 block = (b, 16-query tile), 8 waves.
// ---------------------------------------------------------------------------
__global__ __launch_bounds__(256)
void attn_kernel(const float* __restrict__ x,
                 const __bf16* __restrict__ qa,
                 const __bf16* __restrict__ ker,
                 const __bf16* __restrict__ kei,
                 const __bf16* __restrict__ vbuf,
                 const float* __restrict__ gamma,
                 float* __restrict__ out) {
  __shared__ __align__(16) __bf16 kt_er[32][16];   // staged key rows (1KB)
  __shared__ __align__(16) __bf16 kt_ei[32][16];
  __shared__ __align__(16) __bf16 p_lds[16][32];
  __shared__ float red_m[16][16];
  __shared__ float red_s[16][16];
  __shared__ float row_max[16];
  __shared__ float row_inv[16];

  const int tid  = threadIdx.x;
  const int lane = tid & 31;
  const int wave = tid >> 5;
  const int half = lane >> 4;
  const int n    = lane & 15;

  const int blk = blockIdx.x;          // B * 256
  const int b   = blk >> 8;
  const int q0  = (blk & 255) * 16;

  const float g = gamma[0];
  const int qRow = tid & 15;
  const int kk   = (tid >> 4) * 2;     // this thread's 2 keys within a 32-block

  float qv[16];
  {
    const __bf16* qp = qa + ((size_t)(b * L_ + q0 + qRow) << 4);
#pragma unroll
    for (int j = 0; j < 16; ++j) qv[j] = (float)qp[j];
  }

  const __bf16* kerB = ker + (((size_t)b * L_) << 4);
  const __bf16* keiB = kei + (((size_t)b * L_) << 4);

  // ---- pass 1: softmax stats with LDS-staged key blocks
  float m = -3.0e38f, s = 0.0f;
  for (int k0 = 0; k0 < L_; k0 += 32) {
#ifdef USE_ASYNC_LDS
    if (tid < 64) {
      ASYNC_CP16(kerB + (((size_t)k0) << 4) + tid * 8, (__bf16*)kt_er + tid * 8);
    } else if (tid < 128) {
      const int t2 = tid - 64;
      ASYNC_CP16(keiB + (((size_t)k0) << 4) + t2 * 8, (__bf16*)kt_ei + t2 * 8);
    }
    __builtin_amdgcn_s_wait_asynccnt(0);
#else
    if (tid < 128) {
      const int which = tid >> 6;
      const int t2 = tid & 63;
      const __bf16* srcp = (which ? keiB : kerB) + (((size_t)k0) << 4) + t2 * 8;
      __bf16* dstp = (which ? (__bf16*)kt_ei : (__bf16*)kt_er) + t2 * 8;
      *(uint4*)dstp = *(const uint4*)srcp;
    }
#endif
    __syncthreads();
#pragma unroll
    for (int u = 0; u < 2; ++u) {
      const __bf16* erp = &kt_er[kk + u][0];
      const __bf16* eip = &kt_ei[kk + u][0];
      float er = 0.f, ei = 0.f;
#pragma unroll
      for (int j = 0; j < 16; ++j) {
        er += qv[j] * (float)erp[j];
        ei += qv[j] * (float)eip[j];
      }
      const float sc = er * er + ei * ei;
      if (sc > m) { s = s * __expf(m - sc) + 1.0f; m = sc; }
      else        { s += __expf(sc - m); }
    }
    __syncthreads();
  }
  red_m[qRow][tid >> 4] = m;
  red_s[qRow][tid >> 4] = s;
  __syncthreads();
  if (tid < 16) {
    float M = -3.0e38f;
#pragma unroll
    for (int j = 0; j < 16; ++j) M = fmaxf(M, red_m[tid][j]);
    float S = 0.0f;
#pragma unroll
    for (int j = 0; j < 16; ++j) S += red_s[tid][j] * __expf(red_m[tid][j] - M);
    row_max[tid] = M;
    row_inv[tid] = 1.0f / S;
  }
  __syncthreads();

  const float rmax = row_max[qRow];
  const float rinv = row_inv[qRow];

  // ---- pass 2: rebuild P and run P x V^T with WMMA
  const __bf16* vB = vbuf + (size_t)b * CTOT * L_;
  const int cTile0 = wave * 4;

  v8f acc[4];
#pragma unroll
  for (int j = 0; j < 4; ++j) acc[j] = v8f_zero();

  for (int k0 = 0; k0 < L_; k0 += 32) {
#ifdef USE_ASYNC_LDS
    if (tid < 64) {
      ASYNC_CP16(kerB + (((size_t)k0) << 4) + tid * 8, (__bf16*)kt_er + tid * 8);
    } else if (tid < 128) {
      const int t2 = tid - 64;
      ASYNC_CP16(keiB + (((size_t)k0) << 4) + t2 * 8, (__bf16*)kt_ei + t2 * 8);
    }
    __builtin_amdgcn_s_wait_asynccnt(0);
#else
    if (tid < 128) {
      const int which = tid >> 6;
      const int t2 = tid & 63;
      const __bf16* srcp = (which ? keiB : kerB) + (((size_t)k0) << 4) + t2 * 8;
      __bf16* dstp = (which ? (__bf16*)kt_ei : (__bf16*)kt_er) + t2 * 8;
      *(uint4*)dstp = *(const uint4*)srcp;
    }
#endif
    __syncthreads();

#pragma unroll
    for (int u = 0; u < 2; ++u) {
      const __bf16* erp = &kt_er[kk + u][0];
      const __bf16* eip = &kt_ei[kk + u][0];
      float er = 0.f, ei = 0.f;
#pragma unroll
      for (int j = 0; j < 16; ++j) {
        er += qv[j] * (float)erp[j];
        ei += qv[j] * (float)eip[j];
      }
      const float sc = er * er + ei * ei;
      p_lds[qRow][kk + u] = (__bf16)(__expf(sc - rmax) * rinv);
    }
    __syncthreads();

    // A tile: P per-lane (row = n, K = half*16 + e, contiguous)
    const int kbase = half * 16;
    const v16bf a_p = bf_cat(*(const v8bf*)&p_lds[n][kbase],
                             *(const v8bf*)&p_lds[n][kbase + 8]);

    // B tiles: V^T, col = channel, K = k0 + half*16 + e (contiguous)
#pragma unroll
    for (int j = 0; j < 4; ++j) {
      const int c = (cTile0 + j) * 16 + n;
      const __bf16* vp = vB + (size_t)c * L_ + k0 + half * 16;
      __builtin_prefetch(vp + 32, 0, 0);
      const v16bf b_v = bf_cat(*(const v8bf*)vp, *(const v8bf*)(vp + 8));
      acc[j] = WMMA_BF16(a_p, b_v, acc[j]);
    }
    __syncthreads();
  }

  // ---- epilogue: out = gamma * acc + x
#pragma unroll
  for (int j = 0; j < 4; ++j) {
    const int cfull = (cTile0 + j) * 16 + n;
    const int comp  = cfull >> 8;
    const int ch    = cfull & 255;
    const size_t base = (((size_t)comp * B_ + b) * CG_ + ch) * L_;
#pragma unroll
    for (int i = 0; i < 8; ++i) {
      const int q = q0 + i + half * 8;
      out[base + q] = g * acc[j][i] + x[base + q];
    }
  }
}

// ---------------------------------------------------------------------------
// launch
// ---------------------------------------------------------------------------
extern "C" void kernel_launch(void* const* d_in, const int* in_sizes, int n_in,
                              void* d_out, int out_size, void* d_ws, size_t ws_size,
                              hipStream_t stream) {
  const float* x     = (const float*)d_in[0];
  const float* wq    = (const float*)d_in[1];
  const float* wk    = (const float*)d_in[2];
  const float* wv    = (const float*)d_in[3];
  const float* gamma = (const float*)d_in[4];
  float* out = (float*)d_out;

  // workspace: qa 256K | ker 256K | kei 256K | wvb 256K | vbuf 8M | xbT 8M
  char* ws = (char*)d_ws;
  __bf16* qa   = (__bf16*)(ws);
  __bf16* ker  = (__bf16*)(ws + (256 << 10));
  __bf16* kei  = (__bf16*)(ws + (512 << 10));
  __bf16* wvb  = (__bf16*)(ws + (768 << 10));
  __bf16* vbuf = (__bf16*)(ws + (1 << 20));
  __bf16* xbT  = (__bf16*)(ws + (9 << 20));

  xpose_kernel <<<4096, 256, 0, stream>>>(x, xbT);
  wconv_kernel <<<512, 256, 0, stream>>>(wv, wvb);
  qk_pack_kernel<<<256, 256, 0, stream>>>(xbT, wq, wk, qa, ker, kei);
  vgemm_kernel <<<B_ * 16 * 256, 32, 0, stream>>>(xbT, wvb, vbuf);
  attn_kernel  <<<B_ * 256, 256, 0, stream>>>(x, qa, ker, kei, vbuf, gamma, out);
}
